// MambaEncoder_48850958025091
// MI455X (gfx1250) — compile-verified, matrix operand order checked by reading, not run on hardware
//
#include <hip/hip_runtime.h>
#include <hip/hip_bf16.h>

// ---------------------------------------------------------------------------
// MambaEncoder forward for gfx1250 (MI455X).
// GEMMs run on the WMMA pipe (v_wmma_f32_16x16x32_f16, f32 accumulate);
// layernorm / conv / selective-scan / gating stay fp32 VALU.
// Each wave computes a 16(M) x 64(N) strip: one A fragment feeds 4
// back-to-back WMMAs per K-step (4 f32 accumulators in registers).
// All GEMM N sizes are multiples of 64 (x_proj weight is zero-padded
// 80 -> 128 rows in workspace), so the K-loop is fully branch-free.
// ---------------------------------------------------------------------------

typedef __attribute__((ext_vector_type(16))) _Float16 v16h;
typedef __attribute__((ext_vector_type(8)))  float    v8f;

static constexpr int BDIM    = 8;
static constexpr int LSEQ    = 512;
static constexpr int IN_DIM  = 128;
static constexpr int D_MODEL = 256;
static constexpr int D_INNER = 1024;
static constexpr int D_STATE = 32;
static constexpr int D_CONV  = 8;
static constexpr int DT_RANK = 16;
static constexpr int OUT_DIM = 128;
static constexpr int MTOK    = BDIM * LSEQ;   // 4096 token rows
static constexpr int XD_PAD  = 128;           // padded x_dbl width (was 80)

// Epilogue flags for the generic GEMM
static constexpr int EPI_BIAS     = 1;  // += bias[n]
static constexpr int EPI_POS      = 2;  // += extra[(m & posMask) * N + n]
static constexpr int EPI_RESID    = 4;  // += extra[m * N + n]
static constexpr int EPI_SOFTPLUS = 8;  // softplus activation

// ---------------------------------------------------------------------------
// f16 fragment builders.  Per ISA 7.12.2 (16-bit A-matrix 16x32): lane
// (half|row): row = lane&15 selects the M (N for B) line, half = lane>>4
// selects the K sub-block.  Elements 0..7 -> K = k0 + half*8 + e,
// elements 8..15 -> K = k0 + 16 + half*8 + (e-8).
// ---------------------------------------------------------------------------
__device__ inline v16h cvt_frag(float4 x0, float4 x1, float4 x2, float4 x3) {
  v16h f;
  f[0]  = (_Float16)x0.x; f[1]  = (_Float16)x0.y; f[2]  = (_Float16)x0.z; f[3]  = (_Float16)x0.w;
  f[4]  = (_Float16)x1.x; f[5]  = (_Float16)x1.y; f[6]  = (_Float16)x1.z; f[7]  = (_Float16)x1.w;
  f[8]  = (_Float16)x2.x; f[9]  = (_Float16)x2.y; f[10] = (_Float16)x2.z; f[11] = (_Float16)x2.w;
  f[12] = (_Float16)x3.x; f[13] = (_Float16)x3.y; f[14] = (_Float16)x3.z; f[15] = (_Float16)x3.w;
  return f;
}

// Full 32-wide K step: p = rowBase + k0 + kh (32B aligned), pair at +16.
__device__ inline v16h load_frag_full(const float* p) {
  const float4* a = reinterpret_cast<const float4*>(p);
  const float4* b = reinterpret_cast<const float4*>(p + 16);
  return cvt_frag(a[0], a[1], b[0], b[1]);
}

// Guarded tail step (K not a multiple of 32, e.g. K == DT_RANK == 16).
__device__ inline v16h load_frag_tail(const float* rowBase, int K, int k0, int kh) {
  v16h f;
#pragma unroll
  for (int e = 0; e < 8; ++e) {
    const int k1 = k0 + kh + e;
    const int k2 = k0 + 16 + kh + e;
    f[e]     = (k1 < K) ? (_Float16)rowBase[k1] : (_Float16)0.0f;
    f[e + 8] = (k2 < K) ? (_Float16)rowBase[k2] : (_Float16)0.0f;
  }
  return f;
}

// ---------------------------------------------------------------------------
// Generic GEMM:  C[M,N] = A[M,K(lda)] * W[N,K]^T  (+ epilogue)
// Requires N % 64 == 0 and M % 128 == 0.
// Block = 256 threads = 8 waves stacked in M; wave tile = 16(M) x 64(N);
// block tile = 128(M) x 64(N).  grid = (N/64, M/128).
// ---------------------------------------------------------------------------
template <int EPI>
__global__ __launch_bounds__(256) void gemm_wmma_kernel(
    const float* __restrict__ A, const float* __restrict__ W,
    const float* __restrict__ bias, const float* __restrict__ extra,
    float* __restrict__ C, int M, int N, int K, int lda, int posMask) {
  const int lane  = threadIdx.x & 31;
  const int wave  = threadIdx.x >> 5;
  const int tileM = blockIdx.y * 128 + wave * 16;
  const int tileN = blockIdx.x * 64;

  const int row = lane & 15;                 // M line for A, N line for B
  const int kh  = (lane >> 4) * 8;           // K sub-block select
  const float* Arow = A + (size_t)(tileM + row) * (size_t)lda;
  const float* Wrow = W + (size_t)(tileN + row) * (size_t)K;

  v8f acc[4] = {{}, {}, {}, {}};
  const int kmain = K & ~31;

  for (int k0 = 0; k0 < kmain; k0 += 32) {   // branch-free steady state
    __builtin_prefetch(Arow + k0 + 64, 0, 1);            // speculative
    const v16h af = load_frag_full(Arow + k0 + kh);
#pragma unroll
    for (int j = 0; j < 4; ++j) {
      const v16h bf = load_frag_full(Wrow + (size_t)j * 16 * (size_t)K + k0 + kh);
      acc[j] = __builtin_amdgcn_wmma_f32_16x16x32_f16(
          false, af, false, bf, (short)0, acc[j], false, false);
    }
  }
  if (kmain < K) {                           // single guarded tail (K%32 != 0)
    const v16h af = load_frag_tail(Arow, K, kmain, kh);
#pragma unroll
    for (int j = 0; j < 4; ++j) {
      const v16h bf = load_frag_tail(Wrow + (size_t)j * 16 * (size_t)K, K, kmain, kh);
      acc[j] = __builtin_amdgcn_wmma_f32_16x16x32_f16(
          false, af, false, bf, (short)0, acc[j], false, false);
    }
  }

  // C/D layout: lane n = lane&15, rows m = (lane>>4)*8 + r  (r = VGPR index)
  const int mbase = tileM + (lane >> 4) * 8;
#pragma unroll
  for (int j = 0; j < 4; ++j) {
    const int n = tileN + j * 16 + (lane & 15);
    float bn = 0.0f;
    if (EPI & EPI_BIAS) bn = bias[n];
#pragma unroll
    for (int r = 0; r < 8; ++r) {
      const int m = mbase + r;
      float v = acc[j][r] + bn;
      if (EPI & EPI_POS)   v += extra[(size_t)(m & posMask) * (size_t)N + n];
      if (EPI & EPI_RESID) v += extra[(size_t)m * (size_t)N + n];
      if (EPI & EPI_SOFTPLUS) v = (v > 20.0f) ? v : __logf(1.0f + __expf(v));
      C[(size_t)m * (size_t)N + n] = v;
    }
  }
}

// ---------------------------------------------------------------------------
// Zero-pad x_proj_w (80 x 1024) -> wpad (128 x 1024).
// ---------------------------------------------------------------------------
__global__ __launch_bounds__(256) void pad_xproj_kernel(
    const float* __restrict__ w, float* __restrict__ wp) {
  const int idx = blockIdx.x * 256 + threadIdx.x;   // XD_PAD * D_INNER
  const int n = idx >> 10;
  const int k = idx & (D_INNER - 1);
  wp[idx] = (n < DT_RANK + 2 * D_STATE) ? w[n * D_INNER + k] : 0.0f;
}

// ---------------------------------------------------------------------------
// LayerNorm over D_MODEL=256: one wave32 per row, 8 rows per 256-thread block.
// ---------------------------------------------------------------------------
__global__ __launch_bounds__(256) void layernorm_kernel(
    const float* __restrict__ h, const float* __restrict__ w,
    const float* __restrict__ b, float* __restrict__ xn) {
  const int lane = threadIdx.x & 31;
  const int wave = threadIdx.x >> 5;
  const int m = blockIdx.x * 8 + wave;
  const float* rowp = h + (size_t)m * D_MODEL;

  float vals[8];
  float s = 0.0f;
#pragma unroll
  for (int i = 0; i < 8; ++i) { vals[i] = rowp[lane + i * 32]; s += vals[i]; }
#pragma unroll
  for (int off = 16; off > 0; off >>= 1) s += __shfl_xor(s, off, 32);
  const float mu = s * (1.0f / 256.0f);

  float vs = 0.0f;
#pragma unroll
  for (int i = 0; i < 8; ++i) { float d = vals[i] - mu; vs += d * d; }
#pragma unroll
  for (int off = 16; off > 0; off >>= 1) vs += __shfl_xor(vs, off, 32);
  const float rstd = rsqrtf(vs * (1.0f / 256.0f) + 1e-5f);

  float* outp = xn + (size_t)m * D_MODEL;
#pragma unroll
  for (int i = 0; i < 8; ++i) {
    const int n = lane + i * 32;
    outp[n] = (vals[i] - mu) * rstd * w[n] + b[n];
  }
}

// ---------------------------------------------------------------------------
// Depthwise causal conv (kernel 8) over L + bias + SiLU.
// u lives in xz[:, 0:D_INNER] (row stride 2*D_INNER).
// ---------------------------------------------------------------------------
__global__ __launch_bounds__(256) void conv_silu_kernel(
    const float* __restrict__ xz, const float* __restrict__ cw,
    const float* __restrict__ cb, float* __restrict__ uc) {
  const int idx = blockIdx.x * blockDim.x + threadIdx.x;   // MTOK*D_INNER
  const int d = idx & (D_INNER - 1);
  const int m = idx >> 10;            // b*LSEQ + l
  const int l = m & (LSEQ - 1);
  float acc = cb[d];
#pragma unroll
  for (int j = 0; j < D_CONV; ++j) {
    const int lp = l - (D_CONV - 1) + j;
    if (lp >= 0)
      acc = fmaf(cw[d * D_CONV + j],
                 xz[(size_t)(m - (D_CONV - 1) + j) * (2 * D_INNER) + d], acc);
  }
  const float sig = 1.0f / (1.0f + __expf(-acc));
  uc[idx] = acc * sig;
}

// ---------------------------------------------------------------------------
// Selective scan, fused with D-skip and silu(z) gating.
// One block = fixed batch b, 256 contiguous channels d; 32-state recurrence
// held in registers; per-timestep B/C rows broadcast through LDS.
// ---------------------------------------------------------------------------
__global__ __launch_bounds__(256) void scan_kernel(
    const float* __restrict__ xdbl,   // [MTOK, XD_PAD]  (dt_in | B | C | pad)
    const float* __restrict__ dtb,    // [MTOK, D_INNER] softplus(dt)
    const float* __restrict__ uc,     // [MTOK, D_INNER]
    const float* __restrict__ xz,     // [MTOK, 2*D_INNER] (z = cols 1024..)
    const float* __restrict__ A_log,  // [D_INNER, D_STATE]
    const float* __restrict__ Dp,     // [D_INNER]
    float* __restrict__ ygb) {        // [MTOK, D_INNER]
  const int b     = blockIdx.x >> 2;            // 8 batches x 4 channel chunks
  const int dbase = (blockIdx.x & 3) * 256;
  const int d     = dbase + threadIdx.x;

  __shared__ float Bs[D_STATE];
  __shared__ float Cs[D_STATE];

  float A[D_STATE], h[D_STATE];
#pragma unroll
  for (int s = 0; s < D_STATE; ++s) {
    A[s] = -__expf(A_log[(size_t)d * D_STATE + s]);
    h[s] = 0.0f;
  }
  const float Dd = Dp[d];

  for (int l = 0; l < LSEQ; ++l) {
    const int m = b * LSEQ + l;
    __syncthreads();                       // protect LDS WAR from prev step
    if (threadIdx.x < 2 * D_STATE) {
      const float v = xdbl[(size_t)m * XD_PAD + DT_RANK + threadIdx.x];
      if (threadIdx.x < D_STATE) Bs[threadIdx.x] = v;
      else                       Cs[threadIdx.x - D_STATE] = v;
    }
    __syncthreads();

    const float dt  = dtb[(size_t)m * D_INNER + d];
    const float u   = uc [(size_t)m * D_INNER + d];
    const float z   = xz [(size_t)m * (2 * D_INNER) + D_INNER + d];
    const float dtu = dt * u;
    float y = 0.0f;
#pragma unroll
    for (int s = 0; s < D_STATE; ++s) {
      const float dA = __expf(dt * A[s]);
      h[s] = fmaf(dA, h[s], dtu * Bs[s]);
      y = fmaf(h[s], Cs[s], y);
    }
    y += u * Dd;
    const float sig = 1.0f / (1.0f + __expf(-z));
    ygb[(size_t)m * D_INNER + d] = y * (z * sig);
  }
}

// ---------------------------------------------------------------------------
// Host-side orchestration.  Input order = setup_inputs() flattening:
//   0:x 1:in_w 2:in_b 3:pos 4:fc_w 5:fc_b, then per layer (11 each):
//   ln_w ln_b in_proj_w conv_w conv_b x_proj_w dt_proj_w dt_proj_b A_log D out_proj_w
// ---------------------------------------------------------------------------
extern "C" void kernel_launch(void* const* d_in, const int* in_sizes, int n_in,
                              void* d_out, int out_size, void* d_ws, size_t ws_size,
                              hipStream_t stream) {
  (void)in_sizes; (void)n_in; (void)out_size; (void)ws_size;

  const float* x    = (const float*)d_in[0];
  const float* in_w = (const float*)d_in[1];
  const float* in_b = (const float*)d_in[2];
  const float* pos  = (const float*)d_in[3];
  const float* fc_w = (const float*)d_in[4];
  const float* fc_b = (const float*)d_in[5];

  float* ws   = (float*)d_ws;
  float* hbuf = ws;                                   // [MTOK, 256]
  float* xn   = hbuf + (size_t)MTOK * D_MODEL;        // [MTOK, 256]
  float* xz   = xn   + (size_t)MTOK * D_MODEL;        // [MTOK, 2048]
  float* uc   = xz   + (size_t)MTOK * 2 * D_INNER;    // [MTOK, 1024]
  float* xdbl = uc   + (size_t)MTOK * D_INNER;        // [MTOK, 128]
  float* dtb  = xdbl + (size_t)MTOK * XD_PAD;         // [MTOK, 1024]
  float* ygb  = dtb  + (size_t)MTOK * D_INNER;        // [MTOK, 1024]
  float* wpad = ygb  + (size_t)MTOK * D_INNER;        // [128, 1024]

  const dim3 blk(256);
  const int gy = MTOK / 128;   // 32 block-rows of M

  // Embedding: h = x @ in_w.T + in_b + pos[l]
  gemm_wmma_kernel<EPI_BIAS | EPI_POS>
      <<<dim3(D_MODEL / 64, gy), blk, 0, stream>>>(
          x, in_w, in_b, pos, hbuf, MTOK, D_MODEL, IN_DIM, IN_DIM, LSEQ - 1);

  for (int ly = 0; ly < 4; ++ly) {
    void* const* L = d_in + 6 + ly * 11;
    const float* ln_w      = (const float*)L[0];
    const float* ln_b      = (const float*)L[1];
    const float* in_proj_w = (const float*)L[2];
    const float* conv_w    = (const float*)L[3];
    const float* conv_b    = (const float*)L[4];
    const float* x_proj_w  = (const float*)L[5];
    const float* dt_proj_w = (const float*)L[6];
    const float* dt_proj_b = (const float*)L[7];
    const float* A_log     = (const float*)L[8];
    const float* Dp        = (const float*)L[9];
    const float* out_proj_w= (const float*)L[10];

    layernorm_kernel<<<MTOK / 8, blk, 0, stream>>>(hbuf, ln_w, ln_b, xn);

    // xz = xn @ in_proj_w.T   (u | z)
    gemm_wmma_kernel<0><<<dim3(2 * D_INNER / 64, gy), blk, 0, stream>>>(
        xn, in_proj_w, nullptr, nullptr, xz, MTOK, 2 * D_INNER, D_MODEL, D_MODEL, 0);

    // depthwise causal conv + SiLU
    conv_silu_kernel<<<(MTOK * D_INNER) / 256, blk, 0, stream>>>(xz, conv_w, conv_b, uc);

    // zero-pad x_proj_w (80x1024 -> 128x1024) so x_proj GEMM has N%64==0
    pad_xproj_kernel<<<(XD_PAD * D_INNER) / 256, blk, 0, stream>>>(x_proj_w, wpad);

    // x_dbl = u_c @ wpad.T   (N=128, cols 80..127 are dead)
    gemm_wmma_kernel<0><<<dim3(XD_PAD / 64, gy), blk, 0, stream>>>(
        uc, wpad, nullptr, nullptr, xdbl, MTOK, XD_PAD, D_INNER, D_INNER, 0);

    // dt = softplus(x_dbl[:, :16] @ dt_proj_w.T + dt_proj_b)  (K=16, lda=128)
    gemm_wmma_kernel<EPI_BIAS | EPI_SOFTPLUS>
        <<<dim3(D_INNER / 64, gy), blk, 0, stream>>>(
            xdbl, dt_proj_w, dt_proj_b, nullptr, dtb, MTOK, D_INNER, DT_RANK, XD_PAD, 0);

    // selective scan + D-skip + silu(z) gating
    scan_kernel<<<BDIM * 4, blk, 0, stream>>>(xdbl, dtb, uc, xz, A_log, Dp, ygb);

    // h = h + yg @ out_proj_w.T   (residual read/write same element: safe)
    gemm_wmma_kernel<EPI_RESID><<<dim3(D_MODEL / 64, gy), blk, 0, stream>>>(
        ygb, out_proj_w, nullptr, hbuf, hbuf, MTOK, D_MODEL, D_INNER, D_INNER, 0);
  }

  // out = h @ fc_w.T + fc_b
  gemm_wmma_kernel<EPI_BIAS><<<dim3(OUT_DIM / 64, gy), blk, 0, stream>>>(
      hbuf, fc_w, fc_b, nullptr, (float*)d_out, MTOK, OUT_DIM, D_MODEL, D_MODEL, 0);
}
